// NaiveGate_54211077210522
// MI455X (gfx1250) — matrix operand order repaired and checked by reading, not compiled, not run on hardware
//
#include <hip/hip_runtime.h>
#include <hip/hip_bf16.h>

// Problem constants (from reference)
#define D_MODEL   2048
#define N_EXPERT  64
#define KSTEP     32
#define WAVES     8
#define ROWS_PER_WAVE  16
#define ROWS_PER_BLOCK (WAVES * ROWS_PER_WAVE)   // 128
#define WROW      2064                            // padded LDS row stride (bf16 elems) -> 4128 B
#define WBYTES    (N_EXPERT * WROW * 2)           // 264192 B of LDS for W
#define LSTR      68                              // logits row stride (floats)
#define SMEM_BYTES (WBYTES + WAVES * ROWS_PER_WAVE * LSTR * 4)  // ~299 KB < 320 KB WGP LDS

typedef __attribute__((ext_vector_type(16))) __bf16 v16bf;
typedef __attribute__((ext_vector_type(8)))  float  v8f;
typedef __attribute__((ext_vector_type(4)))  float  v4f;

// Async global->LDS copy of 16 bytes per lane (CDNA5 GLOBAL_LOAD_ASYNC_TO_LDS_B128).
// VDST operand = VGPR holding LDS byte offset (added to the LDS base and the instruction offset),
// VADDR = VGPR pair with 64-bit global address. Tracked by ASYNCcnt.
__device__ __forceinline__ void async_copy_b128(void* lds_ptr, const void* gptr) {
    unsigned lds_off = (unsigned)(size_t)lds_ptr;          // low 32 bits of generic = LDS offset
    unsigned long long ga = (unsigned long long)(size_t)gptr;
    asm volatile("global_load_async_to_lds_b128 %0, %1, off"
                 :: "v"(lds_off), "v"(ga) : "memory");
}

__device__ __forceinline__ void wait_asynccnt0() {
#if defined(__has_builtin) && __has_builtin(__builtin_amdgcn_s_wait_asynccnt)
    __builtin_amdgcn_s_wait_asynccnt(0);
#else
    asm volatile("s_wait_asynccnt 0x0" ::: "memory");
#endif
}

// Kernel 0: convert W [64,2048] f32 -> bf16 in workspace (lives in L2 thereafter)
__global__ __launch_bounds__(256) void moe_w_to_bf16(const float* __restrict__ W,
                                                     __bf16* __restrict__ Wb) {
    int i = blockIdx.x * blockDim.x + threadIdx.x;
    if (i < N_EXPERT * D_MODEL) Wb[i] = (__bf16)W[i];
}

// Kernel 1: gate GEMM (bf16 WMMA, f32 accum) + top-2 + softmax + dense scatter
__global__ __launch_bounds__(256) void moe_gate_kernel(const float* __restrict__ inp,
                                                       const __bf16* __restrict__ Wb,
                                                       const float* __restrict__ bias,
                                                       float* __restrict__ out) {
    extern __shared__ __align__(32) char smem[];
    __bf16* wlds = (__bf16*)smem;

    const int tid  = threadIdx.x;
    const int lane = tid & 31;
    const int wave = tid >> 5;
    const int n    = lane & 15;   // column within 16-wide tile (expert lane / A row)
    const int half = lane >> 4;   // which half of the wave

    // ---- Stage bf16 W into LDS (padded row stride) via async global->LDS DMA ----
    {
        const char* src = (const char*)Wb;            // 256 chunks of 16B per expert row
        for (int i = 0; i < (N_EXPERT * D_MODEL * 2 / 16) / 256; ++i) {
            int c  = tid + 256 * i;                   // chunk id
            int r  = c >> 8;                          // expert row
            int co = c & 255;                         // 16B chunk within row
            async_copy_b128(smem + r * (WROW * 2) + co * 16, src + c * 16);
        }
        wait_asynccnt0();
    }
    __syncthreads();

    const int row0 = (blockIdx.x * WAVES + wave) * ROWS_PER_WAVE;

    v8f acc0 = {}, acc1 = {}, acc2 = {}, acc3 = {};

    // A addressing (contiguous K order, matching B):
    // lanes 0-15 (half=0): K 0..15 in v0-7
    // lanes 16-31 (half=1): K 16..31 in v0-7
    const float* abase = inp + (size_t)(row0 + n) * D_MODEL + half * 16;

    for (int k0 = 0; k0 < D_MODEL; k0 += KSTEP) {
        __builtin_prefetch(abase + k0 + 8 * KSTEP, 0, 3);   // WGP-scope prefetch on A stream

        v4f a0 = *(const v4f*)(abase + k0);
        v4f a1 = *(const v4f*)(abase + k0 + 4);
        v4f a2 = *(const v4f*)(abase + k0 + 8);
        v4f a3 = *(const v4f*)(abase + k0 + 12);
        v16bf af;
        af[0]  = (__bf16)a0[0]; af[1]  = (__bf16)a0[1]; af[2]  = (__bf16)a0[2]; af[3]  = (__bf16)a0[3];
        af[4]  = (__bf16)a1[0]; af[5]  = (__bf16)a1[1]; af[6]  = (__bf16)a1[2]; af[7]  = (__bf16)a1[3];
        af[8]  = (__bf16)a2[0]; af[9]  = (__bf16)a2[1]; af[10] = (__bf16)a2[2]; af[11] = (__bf16)a2[3];
        af[12] = (__bf16)a3[0]; af[13] = (__bf16)a3[1]; af[14] = (__bf16)a3[2]; af[15] = (__bf16)a3[3];

        // B in contiguous K order (same as A): lane = column (expert), lanes 0-15 hold K 0..15,
        // lanes 16-31 hold K 16..31 -> one contiguous 16-elem bf16 run per lane.
        const __bf16* bbase = wlds + k0 + half * 16;
        v16bf b0, b1, b2, b3;
        {
            const char* p0 = (const char*)(bbase + (size_t)(n +  0) * WROW);
            const char* p1 = (const char*)(bbase + (size_t)(n + 16) * WROW);
            const char* p2 = (const char*)(bbase + (size_t)(n + 32) * WROW);
            const char* p3 = (const char*)(bbase + (size_t)(n + 48) * WROW);
            ((uint4*)&b0)[0] = ((const uint4*)p0)[0]; ((uint4*)&b0)[1] = ((const uint4*)p0)[1];
            ((uint4*)&b1)[0] = ((const uint4*)p1)[0]; ((uint4*)&b1)[1] = ((const uint4*)p1)[1];
            ((uint4*)&b2)[0] = ((const uint4*)p2)[0]; ((uint4*)&b2)[1] = ((const uint4*)p2)[1];
            ((uint4*)&b3)[0] = ((const uint4*)p3)[0]; ((uint4*)&b3)[1] = ((const uint4*)p3)[1];
        }

        acc0 = __builtin_amdgcn_wmma_f32_16x16x32_bf16(false, af, false, b0, (short)0, acc0, false, false);
        acc1 = __builtin_amdgcn_wmma_f32_16x16x32_bf16(false, af, false, b1, (short)0, acc1, false, false);
        acc2 = __builtin_amdgcn_wmma_f32_16x16x32_bf16(false, af, false, b2, (short)0, acc2, false, false);
        acc3 = __builtin_amdgcn_wmma_f32_16x16x32_bf16(false, af, false, b3, (short)0, acc3, false, false);
    }

    // ---- accumulators + bias -> per-wave LDS logits tile [16][LSTR] ----
    float* lg = (float*)(smem + WBYTES) + wave * (ROWS_PER_WAVE * LSTR);
    {
        float bb0 = bias[ 0 + n], bb1 = bias[16 + n], bb2 = bias[32 + n], bb3 = bias[48 + n];
#pragma unroll
        for (int v = 0; v < 8; ++v) {
            int m = v + half * 8;                 // C/D layout: VGPR v holds row v (+8 for hi half)
            float* lrow = lg + m * LSTR;
            lrow[ 0 + n] = acc0[v] + bb0;
            lrow[16 + n] = acc1[v] + bb1;
            lrow[32 + n] = acc2[v] + bb2;
            lrow[48 + n] = acc3[v] + bb3;
        }
    }
    __syncthreads();

    // ---- per-row top-2, softmax over the 2 logits, dense scatter (writes whole row) ----
    if (lane < ROWS_PER_WAVE) {
        const float* lrow = lg + lane * LSTR;
        float v1 = -INFINITY, v2 = -INFINITY;
        int i1 = 0, i2 = 0;
        for (int e = 0; e < N_EXPERT; ++e) {
            float v = lrow[e];
            if (v > v1)      { v2 = v1; i2 = i1; v1 = v; i1 = e; }
            else if (v > v2) { v2 = v;  i2 = e; }
        }
        float p1 = 1.0f / (1.0f + __expf(v2 - v1));  // softmax of [v1, v2], v1 = max
        float p2 = 1.0f - p1;
        float* orow = out + (size_t)(row0 + lane) * N_EXPERT;
        for (int e = 0; e < N_EXPERT; ++e) {
            float o = (e == i1) ? p1 : ((e == i2) ? p2 : 0.0f);
            orow[e] = o;
        }
    }
}

extern "C" void kernel_launch(void* const* d_in, const int* in_sizes, int n_in,
                              void* d_out, int out_size, void* d_ws, size_t ws_size,
                              hipStream_t stream) {
    const float* inp  = (const float*)d_in[0];   // [N, 2048] f32
    const float* W    = (const float*)d_in[1];   // [64, 2048] f32
    const float* bias = (const float*)d_in[2];   // [64] f32
    float* out = (float*)d_out;                  // [N, 64] f32
    __bf16* Wb = (__bf16*)d_ws;                  // 256 KB bf16 copy of W

    int n_rows = in_sizes[0] / D_MODEL;          // 32768

    (void)hipFuncSetAttribute((const void*)moe_gate_kernel,
                              hipFuncAttributeMaxDynamicSharedMemorySize, SMEM_BYTES);

    moe_w_to_bf16<<<(N_EXPERT * D_MODEL + 255) / 256, 256, 0, stream>>>(W, Wb);
    moe_gate_kernel<<<n_rows / ROWS_PER_BLOCK, 256, SMEM_BYTES, stream>>>(inp, Wb, bias, out);
}
